// AudioSelfAttention_77902116815075
// MI455X (gfx1250) — matrix operand order repaired and checked by reading, not compiled, not run on hardware
//
#include <hip/hip_runtime.h>
#include <hip/hip_bf16.h>
#include <stdint.h>

// ---------------------------------------------------------------------------
// AudioSelfAttention for MI455X (gfx1250, wave32, WMMA bf16 path)
//   y = proj( softmax( (xWq)(xWk)^T / sqrt(D) ) (xWv) )
// B=2 T=2048 C=1024 H=16 D=64
// ---------------------------------------------------------------------------

#define N_HEADS 16
#define C_EMBD  1024
#define BATCH   2
#define SEQ     2048
#define HDIM    64
#define MROWS   (BATCH * SEQ)     // 4096
#define C3      (3 * C_EMBD)      // 3072

typedef __attribute__((ext_vector_type(16))) __bf16         v16bf;
typedef __attribute__((ext_vector_type(8)))  float          v8f;
typedef __attribute__((ext_vector_type(8)))  unsigned short u16x8;
typedef __attribute__((ext_vector_type(16))) unsigned short u16x16;

static __device__ __forceinline__ unsigned short f2bf(float f) {
    unsigned int u = __float_as_uint(f);
    u += 0x7FFFu + ((u >> 16) & 1u);          // round-to-nearest-even
    return (unsigned short)(u >> 16);
}

static __device__ __forceinline__ v8f vzero8() {
    v8f z = {0.f, 0.f, 0.f, 0.f, 0.f, 0.f, 0.f, 0.f};
    return z;
}

// Two 16-byte loads -> one 16-element bf16 fragment (element order p0[0..7], p1[0..7])
static __device__ __forceinline__ v16bf frag2(const unsigned short* p0,
                                              const unsigned short* p1) {
    u16x8 lo = *reinterpret_cast<const u16x8*>(p0);
    u16x8 hi = *reinterpret_cast<const u16x8*>(p1);
    u16x16 u = __builtin_shufflevector(lo, hi, 0,1,2,3,4,5,6,7,8,9,10,11,12,13,14,15);
    return __builtin_bit_cast(v16bf, u);
}
static __device__ __forceinline__ v16bf frag1(const unsigned short* p) {
    return frag2(p, p + 8);
}

static __device__ __forceinline__ v8f wmma_bf16(v16bf a, v16bf b, v8f c) {
    // D = A(16x32 bf16) x B(32x16 bf16) + C(16x16 f32)
    return __builtin_amdgcn_wmma_f32_16x16x32_bf16(false, a, false, b,
                                                   (short)0, c, false, false);
}

// ---------------------------------------------------------------------------
// Packing kernels: f32 -> bf16 (weights transposed so B-fragments are
// contiguous 32-byte per-lane loads)
// ---------------------------------------------------------------------------
__global__ void pack_cast(const float* __restrict__ src,
                          unsigned short* __restrict__ dst, int n) {
    int i = blockIdx.x * 256 + threadIdx.x;
    if (i < n) dst[i] = f2bf(src[i]);
}

// dst[n*K + k] = bf16(src[k*N + n])   (K x N  ->  N x K transposed)
__global__ void pack_wT(const float* __restrict__ src,
                        unsigned short* __restrict__ dst, int K, int N) {
    int i = blockIdx.x * 256 + threadIdx.x;
    if (i < K * N) {
        int n = i / K, k = i - n * K;
        dst[i] = f2bf(src[(size_t)k * N + n]);
    }
}

// ---------------------------------------------------------------------------
// QKV GEMM: qkv[m, n] = xb[m, :] . W_attn[:, n] + b_attn[n]
// Each wave: 64x32 tile. Ping-pong double-buffered K loop (no register
// rotation moves): 16 WMMA per unrolled 64-K iteration.
// Epilogue scatters into Q [BH,T,D] (pre-scaled by 1/sqrt(D)), K [BH,T,D],
// V transposed [BH,D,T].
// ---------------------------------------------------------------------------
__global__ __launch_bounds__(256) void qkv_gemm(
    const unsigned short* __restrict__ xb,    // [4096][1024] bf16
    const unsigned short* __restrict__ WaT,   // [3072][1024] bf16 (W_attn^T)
    const float* __restrict__ b_attn,         // [3072]
    unsigned short* __restrict__ Qb,          // [32][2048][64]
    unsigned short* __restrict__ Kb,          // [32][2048][64]
    unsigned short* __restrict__ Vt)          // [32][64][2048]
{
    const int gw   = blockIdx.x * 8 + (threadIdx.x >> 5);
    const int m0   = (gw / (C3 / 32)) * 64;
    const int n0   = (gw % (C3 / 32)) * 32;
    const int lane = threadIdx.x & 31;
    const int lr   = lane & 15;
    const int hh   = lane >> 4;

    v8f acc[2][4];
    #pragma unroll
    for (int nt = 0; nt < 2; ++nt)
        #pragma unroll
        for (int mt = 0; mt < 4; ++mt) acc[nt][mt] = vzero8();

    const unsigned short* brow0 = WaT + (size_t)(n0 + lr) * C_EMBD + 16 * hh;
    const unsigned short* brow1 = WaT + (size_t)(n0 + 16 + lr) * C_EMBD + 16 * hh;
    const unsigned short* arow[4];
    #pragma unroll
    for (int mt = 0; mt < 4; ++mt)
        arow[mt] = xb + (size_t)(m0 + mt * 16 + lr) * C_EMBD + 16 * hh;

    v16bf bA0, bA1, aA[4];                    // ping
    v16bf bB0, bB1, aB[4];                    // pong

    auto loadset = [&](v16bf& b0v, v16bf& b1v, v16bf (&av)[4], int k) {
        b0v = frag1(brow0 + k);
        b1v = frag1(brow1 + k);
        #pragma unroll
        for (int mt = 0; mt < 4; ++mt)
            av[mt] = frag2(arow[mt] + k, arow[mt] + k + 8);
    };
    auto mulacc = [&](const v16bf& b0v, const v16bf& b1v, const v16bf (&av)[4]) {
        #pragma unroll
        for (int mt = 0; mt < 4; ++mt) {
            acc[0][mt] = wmma_bf16(av[mt], b0v, acc[0][mt]);
            acc[1][mt] = wmma_bf16(av[mt], b1v, acc[1][mt]);
        }
    };

    loadset(bA0, bA1, aA, 0);
    for (int k0 = 0; k0 < C_EMBD - 64; k0 += 64) {
        loadset(bB0, bB1, aB, k0 + 32);
        mulacc(bA0, bA1, aA);
        loadset(bA0, bA1, aA, k0 + 64);
        mulacc(bB0, bB1, aB);
    }
    loadset(bB0, bB1, aB, C_EMBD - 32);
    mulacc(bA0, bA1, aA);
    mulacc(bB0, bB1, aB);

    #pragma unroll
    for (int nt = 0; nt < 2; ++nt) {
        const int   ncol = n0 + nt * 16 + lr;
        const float bias = b_attn[ncol];
        const int   sec  = ncol >> 10;          // 0=Q 1=K 2=V
        const int   c    = ncol & (C_EMBD - 1);
        const int   head = c >> 6;
        const int   d    = c & (HDIM - 1);
        #pragma unroll
        for (int mt = 0; mt < 4; ++mt) {
            #pragma unroll
            for (int r = 0; r < 8; ++r) {
                const int m  = m0 + mt * 16 + r + 8 * hh;   // C layout: row = r+8h
                const int b  = m >> 11;
                const int t  = m & (SEQ - 1);
                const int bh = b * N_HEADS + head;
                const float v = acc[nt][mt][r] + bias;
                if (sec == 0)
                    Qb[((size_t)bh * SEQ + t) * HDIM + d] = f2bf(v * 0.125f); // 1/sqrt(64)
                else if (sec == 1)
                    Kb[((size_t)bh * SEQ + t) * HDIM + d] = f2bf(v);
                else
                    Vt[((size_t)bh * HDIM + d) * SEQ + t] = f2bf(v);
            }
        }
    }
}

// ---------------------------------------------------------------------------
// Flash attention (transposed): per wave, one 16-query tile of one (b,head).
// S^T = K . Q^T  (C-tile col = query -> softmax stats need one shfl_xor(16))
// O^T += V^T . P^T (accumulators rescaled by per-lane scalar alpha).
// K-fragments ping-pong across 32-key chunks: next chunk's K loads are in
// flight during the current softmax/PV phase. V loads issue at chunk top and
// are hidden under the S-WMMAs + softmax VALU work (XDL co-execution).
// ---------------------------------------------------------------------------
__global__ __launch_bounds__(256) void flash_attn(
    const unsigned short* __restrict__ Qb,
    const unsigned short* __restrict__ Kb,
    const unsigned short* __restrict__ Vt,
    unsigned short* __restrict__ Ob)          // [4096][1024] bf16
{
    const int gw   = blockIdx.x * 8 + (threadIdx.x >> 5);
    const int bh   = gw >> 7;                 // 0..31
    const int t0   = (gw & 127) * 16;         // query tile start
    const int lane = threadIdx.x & 31;
    const int lr   = lane & 15;
    const int hh   = lane >> 4;

    // Q^T B-fragments (lane col = query t0+lr, K = d): loaded once
    const unsigned short* qrow = Qb + ((size_t)bh * SEQ + t0 + lr) * HDIM;
    const v16bf bq0 = frag2(qrow + 8 * hh, qrow + 16 + 8 * hh);        // d in [0,32)
    const v16bf bq1 = frag2(qrow + 32 + 8 * hh, qrow + 48 + 8 * hh);   // d in [32,64)

    const unsigned short* Kbase = Kb + (size_t)bh * SEQ * HDIM;
    const unsigned short* Vbase = Vt + (size_t)bh * HDIM * SEQ;

    float mrow = -1e30f, lrow = 0.f;
    v8f o[4];
    #pragma unroll
    for (int i = 0; i < 4; ++i) o[i] = vzero8();

    v16bf kaA[2][2], kaB[2][2];               // ping-pong K fragments

    auto loadK = [&](v16bf (&ka)[2][2], int key0) {
        const unsigned short* kr0 = Kbase + (size_t)(key0 + lr) * HDIM + 8 * hh;
        const unsigned short* kr1 = Kbase + (size_t)(key0 + 16 + lr) * HDIM + 8 * hh;
        #pragma unroll
        for (int kk = 0; kk < 2; ++kk) {
            ka[0][kk] = frag2(kr0 + kk * 32, kr0 + kk * 32 + 16);
            ka[1][kk] = frag2(kr1 + kk * 32, kr1 + kk * 32 + 16);
        }
    };
    auto loadV = [&](v16bf (&va)[4], int key0) {
        #pragma unroll
        for (int dt = 0; dt < 4; ++dt) {
            const unsigned short* vr = Vbase + (size_t)(dt * 16 + lr) * SEQ + key0 + 16 * hh;
            va[dt] = frag2(vr, vr + 8);
        }
    };
    auto process = [&](const v16bf (&ka)[2][2], const v16bf (&va)[4]) {
        // ---- S^T tiles: s0 = first 16 keys, s1 = next 16 keys of the chunk
        v8f s0 = vzero8(), s1 = vzero8();
        #pragma unroll
        for (int kk = 0; kk < 2; ++kk) {
            const v16bf bq = kk ? bq1 : bq0;
            s0 = wmma_bf16(ka[0][kk], bq, s0);
            s1 = wmma_bf16(ka[1][kk], bq, s1);
        }
        // ---- online softmax over this 32-key chunk (per query = per column)
        float lm = -1e30f;
        #pragma unroll
        for (int r = 0; r < 8; ++r) lm = fmaxf(lm, fmaxf(s0[r], s1[r]));
        lm = fmaxf(lm, __shfl_xor(lm, 16));
        const float mnew  = fmaxf(mrow, lm);
        const float alpha = __expf(mrow - mnew);

        float p0[8], p1[8], lsum = 0.f;
        #pragma unroll
        for (int r = 0; r < 8; ++r) {
            p0[r] = __expf(s0[r] - mnew);
            p1[r] = __expf(s1[r] - mnew);
            lsum += p0[r] + p1[r];
        }
        lsum += __shfl_xor(lsum, 16);
        lrow = lrow * alpha + lsum;
        mrow = mnew;
        #pragma unroll
        for (int i = 0; i < 4; ++i) o[i] *= alpha;

        // ---- build P^T B-fragment (lane col = query, K = key 16h+j);
        // half-wave exchange supplies the other 8 key-rows of the needed tile
        u16x16 pu;
        #pragma unroll
        for (int r = 0; r < 8; ++r) {
            const float q0 = __shfl_xor(p0[r], 16);
            const float q1 = __shfl_xor(p1[r], 16);
            pu[r]     = f2bf(hh ? q1    : p0[r]);
            pu[r + 8] = f2bf(hh ? p1[r] : q0);
        }
        const v16bf pf = __builtin_bit_cast(v16bf, pu);

        // ---- O^T += V^T . P^T  (V fragments already resident)
        #pragma unroll
        for (int dt = 0; dt < 4; ++dt)
            o[dt] = wmma_bf16(va[dt], pf, o[dt]);
    };

    loadK(kaA, 0);
    for (int key0 = 0; key0 < SEQ; key0 += 64) {
        v16bf vaE[4], vaO[4];
        loadV(vaE, key0);
        loadK(kaB, key0 + 32);                // in flight during even chunk
        process(kaA, vaE);
        loadV(vaO, key0 + 32);
        loadK(kaA, key0 + 64);                // last iter overreads into mapped
        process(kaB, vaO);                    // workspace; value never consumed
    }

    // ---- normalize and scatter O^T -> Ob row-major bf16
    const float inv  = 1.0f / lrow;
    const int   b    = bh >> 4;
    const int   head = bh & 15;
    const size_t rowbase = ((size_t)(b * SEQ + t0 + lr)) * C_EMBD + head * HDIM;
    #pragma unroll
    for (int dt = 0; dt < 4; ++dt) {
        #pragma unroll
        for (int r = 0; r < 8; ++r) {
            const int d = dt * 16 + r + 8 * hh;
            Ob[rowbase + d] = f2bf(o[dt][r] * inv);
        }
    }
}

// ---------------------------------------------------------------------------
// Output projection: out[m, n] = Ob[m, :] . W_proj[:, n] + b_proj[n]  (f32 out)
// 64x32 per wave, ping-pong pipelined like qkv_gemm.
// ---------------------------------------------------------------------------
__global__ __launch_bounds__(256) void proj_gemm(
    const unsigned short* __restrict__ Ob,    // [4096][1024] bf16
    const unsigned short* __restrict__ WpT,   // [1024][1024] bf16 (W_proj^T)
    const float* __restrict__ b_proj,
    float* __restrict__ out)                  // [4096][1024] f32
{
    const int gw   = blockIdx.x * 8 + (threadIdx.x >> 5);
    const int m0   = (gw / (C_EMBD / 32)) * 64;
    const int n0   = (gw % (C_EMBD / 32)) * 32;
    const int lane = threadIdx.x & 31;
    const int lr   = lane & 15;
    const int hh   = lane >> 4;

    v8f acc[2][4];
    #pragma unroll
    for (int nt = 0; nt < 2; ++nt)
        #pragma unroll
        for (int mt = 0; mt < 4; ++mt) acc[nt][mt] = vzero8();

    const unsigned short* brow0 = WpT + (size_t)(n0 + lr) * C_EMBD + 16 * hh;
    const unsigned short* brow1 = WpT + (size_t)(n0 + 16 + lr) * C_EMBD + 16 * hh;
    const unsigned short* arow[4];
    #pragma unroll
    for (int mt = 0; mt < 4; ++mt)
        arow[mt] = Ob + (size_t)(m0 + mt * 16 + lr) * C_EMBD + 16 * hh;

    v16bf bA0, bA1, aA[4];
    v16bf bB0, bB1, aB[4];

    auto loadset = [&](v16bf& b0v, v16bf& b1v, v16bf (&av)[4], int k) {
        b0v = frag1(brow0 + k);
        b1v = frag1(brow1 + k);
        #pragma unroll
        for (int mt = 0; mt < 4; ++mt)
            av[mt] = frag2(arow[mt] + k, arow[mt] + k + 8);
    };
    auto mulacc = [&](const v16bf& b0v, const v16bf& b1v, const v16bf (&av)[4]) {
        #pragma unroll
        for (int mt = 0; mt < 4; ++mt) {
            acc[0][mt] = wmma_bf16(av[mt], b0v, acc[0][mt]);
            acc[1][mt] = wmma_bf16(av[mt], b1v, acc[1][mt]);
        }
    };

    loadset(bA0, bA1, aA, 0);
    for (int k0 = 0; k0 < C_EMBD - 64; k0 += 64) {
        loadset(bB0, bB1, aB, k0 + 32);
        mulacc(bA0, bA1, aA);
        loadset(bA0, bA1, aA, k0 + 64);
        mulacc(bB0, bB1, aB);
    }
    loadset(bB0, bB1, aB, C_EMBD - 32);
    mulacc(bA0, bA1, aA);
    mulacc(bB0, bB1, aB);

    #pragma unroll
    for (int nt = 0; nt < 2; ++nt) {
        const int   ncol = n0 + nt * 16 + lr;
        const float bias = b_proj[ncol];
        #pragma unroll
        for (int mt = 0; mt < 4; ++mt) {
            #pragma unroll
            for (int r = 0; r < 8; ++r) {
                const int m = m0 + mt * 16 + r + 8 * hh;
                out[(size_t)m * C_EMBD + ncol] = acc[nt][mt][r] + bias;
            }
        }
    }
}

// ---------------------------------------------------------------------------
// Workspace layout (bytes):   total 48 MB
//   [ 0,  8M) xb     [ 8M, 14M) WaT    [14M, 16M) WpT
//   [16M, 24M) Qb    [24M, 32M) Kb     [32M, 40M) Vt    [40M, 48M) Ob
// ---------------------------------------------------------------------------
extern "C" void kernel_launch(void* const* d_in, const int* in_sizes, int n_in,
                              void* d_out, int out_size, void* d_ws, size_t ws_size,
                              hipStream_t stream) {
    const float* x      = (const float*)d_in[0];
    const float* W_attn = (const float*)d_in[1];
    const float* b_attn = (const float*)d_in[2];
    const float* W_proj = (const float*)d_in[3];
    const float* b_proj = (const float*)d_in[4];

    char* ws = (char*)d_ws;
    const size_t MB = 1024 * 1024;
    unsigned short* xb  = (unsigned short*)(ws);
    unsigned short* WaT = (unsigned short*)(ws + 8 * MB);
    unsigned short* WpT = (unsigned short*)(ws + 14 * MB);
    unsigned short* Qb  = (unsigned short*)(ws + 16 * MB);
    unsigned short* Kb  = (unsigned short*)(ws + 24 * MB);
    unsigned short* Vt  = (unsigned short*)(ws + 32 * MB);
    unsigned short* Ob  = (unsigned short*)(ws + 40 * MB);

    // 1) pack f32 -> bf16 (weights transposed for contiguous B-fragments)
    pack_cast<<<(MROWS * C_EMBD + 255) / 256, 256, 0, stream>>>(x, xb, MROWS * C_EMBD);
    pack_wT<<<(C_EMBD * C3 + 255) / 256, 256, 0, stream>>>(W_attn, WaT, C_EMBD, C3);
    pack_wT<<<(C_EMBD * C_EMBD + 255) / 256, 256, 0, stream>>>(W_proj, WpT, C_EMBD, C_EMBD);

    // 2) QKV projection  (64x32 per wave; 6144 waves -> 768 blocks of 8 waves)
    qkv_gemm<<<(MROWS / 64) * (C3 / 32) / 8, 256, 0, stream>>>(xb, WaT, b_attn, Qb, Kb, Vt);

    // 3) flash attention (one 16-query tile per wave; 4096 waves -> 512 blocks)
    flash_attn<<<(BATCH * N_HEADS * (SEQ / 16)) / 8, 256, 0, stream>>>(Qb, Kb, Vt, Ob);

    // 4) output projection (64x32 per wave; 2048 waves -> 256 blocks)
    proj_gemm<<<(MROWS / 64) * (C_EMBD / 32) / 8, 256, 0, stream>>>(Ob, WpT, b_proj, (float*)d_out);
}
